// DEV_ONE_85521388798276
// MI455X (gfx1250) — matrix operand order repaired and checked by reading, not compiled, not run on hardware
//
#include <hip/hip_runtime.h>
#include <hip/hip_bf16.h>

typedef __bf16 bf16;
typedef __attribute__((ext_vector_type(16))) __bf16 v16bf;
typedef __attribute__((ext_vector_type(8)))  __bf16 v8bf;
typedef __attribute__((ext_vector_type(8)))  float  v8f;

// ---------------- workspace layout (bytes) ----------------
constexpr size_t OFF_QT    = 0;                       // 1024 f32
constexpr size_t OFF_S     = 4096;                    // 64*256 f32
constexpr size_t OFF_H1    = OFF_S     + 65536;       // 64*64 f32
constexpr size_t OFF_H2    = OFF_H1    + 16384;       // 64*32 f32
constexpr size_t OFF_LOWER = OFF_H2    + 8192;        // 64*16384 f32
constexpr size_t OFF_ZTBF  = OFF_LOWER + 4194304;     // 8192*1024 bf16
constexpr size_t OFF_WEMB  = OFF_ZTBF  + 16777216;    // 512*1024 bf16
constexpr size_t OFF_WQKV  = OFF_WEMB  + 1048576;     // 4*1536*512 bf16
constexpr size_t OFF_WOUT  = OFF_WQKV  + 6291456;     // 4*512*512 bf16
constexpr size_t OFF_WFF1  = OFF_WOUT  + 2097152;     // 4*1024*512 bf16
constexpr size_t OFF_WFF2  = OFF_WFF1  + 4194304;     // 4*512*1024 bf16
constexpr size_t OFF_WPRJ  = OFF_WFF2  + 4194304;     // 128*512 bf16
constexpr size_t OFF_T     = OFF_WPRJ  + 131072;      // 8192*512 f32
constexpr size_t OFF_TBF   = OFF_T     + 16777216;    // 8192*512 bf16
constexpr size_t OFF_QKV   = OFF_TBF   + 8388608;     // 8192*1536 f32  (also PRE / final H)
constexpr size_t OFF_PRE   = OFF_QKV;                 // alias (qkv dead by then)
constexpr size_t OFF_ATT   = OFF_QKV   + 50331648;    // 64*8*128*128 f32 (also FFBF)
constexpr size_t OFF_FFBF  = OFF_ATT;                 // alias (att dead by then)
constexpr size_t OFF_CTXBF = OFF_ATT   + 33554432;    // 8192*512 bf16 (also HBF)
constexpr size_t OFF_HP    = OFF_CTXBF + 8388608;     // 8192*128 f32
constexpr size_t OFF_HPBF  = OFF_HP    + 4194304;     // 8192*128 bf16
constexpr size_t OFF_LAMBF = OFF_HPBF  + 2097152;     // 64*128*128 bf16
constexpr size_t OFF_G     = OFF_LAMBF + 2097152;     // 64*128*128 f32
constexpr size_t WS_NEED   = OFF_G     + 4194304;

// ---------------- erfinv (Giles single-precision) ----------------
__device__ __forceinline__ float erfinv_dev(float x) {
    float w = -__logf((1.0f - x) * (1.0f + x));
    float p;
    if (w < 5.0f) {
        w -= 2.5f;
        p = 2.81022636e-08f;
        p = 3.43273939e-07f + p * w;  p = -3.5233877e-06f  + p * w;
        p = -4.39150654e-06f + p * w; p = 0.00021858087f   + p * w;
        p = -0.00125372503f + p * w;  p = -0.00417768164f  + p * w;
        p = 0.246640727f + p * w;     p = 1.50140941f      + p * w;
    } else {
        w = sqrtf(w) - 3.0f;
        p = -0.000200214257f;
        p = 0.000100950558f + p * w;  p = 0.00134934322f   + p * w;
        p = -0.00367342844f + p * w;  p = 0.00573950773f   + p * w;
        p = -0.0076224613f + p * w;   p = 0.00943887047f   + p * w;
        p = 1.00167406f + p * w;      p = 2.83297682f      + p * w;
    }
    return p * x;
}

// q_table[r] = sqrt(2)*erfinv(2*clip((r+1.5)/1024)-1)   (rank r+1, 1-based)
__global__ void build_qtab(float* q) {
    int r = blockIdx.x * 256 + threadIdx.x;
    if (r < 1024) {
        float u = (r + 1.5f) * (1.0f / 1024.0f);
        u = fminf(fmaxf(u, 1e-6f), 1.0f - 1e-6f);
        q[r] = 1.41421356237f * erfinv_dev(2.0f * u - 1.0f);
    }
}

// ---------------- copula: rank -> z (LUT), plus mean/std feats ----------------
__global__ __launch_bounds__(256) void copula_rank(const float* __restrict__ x,
                                                   const float* __restrict__ qtab,
                                                   bf16* __restrict__ ztbf,
                                                   float* __restrict__ sfeat) {
    __shared__ float sx[1024];
    __shared__ float r1[256], r2[256];
    const int blk = blockIdx.x;            // b*128 + d
    const int b = blk >> 7, d = blk & 127;
    const int tid = threadIdx.x;
    for (int i = tid; i < 1024; i += 256)
        sx[i] = x[((long long)b * 1024 + i) * 128 + d];
    __syncthreads();
    const int i0 = tid, i1 = tid + 256, i2 = tid + 512, i3 = tid + 768;
    const float v0 = sx[i0], v1 = sx[i1], v2 = sx[i2], v3 = sx[i3];
    int c0 = 0, c1 = 0, c2 = 0, c3 = 0;
    for (int j = 0; j < 1024; ++j) {
        float w = sx[j];
        c0 += (w < v0) || (w == v0 && j <= i0);
        c1 += (w < v1) || (w == v1 && j <= i1);
        c2 += (w < v2) || (w == v2 && j <= i2);
        c3 += (w < v3) || (w == v3 && j <= i3);
    }
    const float z0 = qtab[c0 - 1], z1 = qtab[c1 - 1], z2 = qtab[c2 - 1], z3 = qtab[c3 - 1];
    bf16* zp = ztbf + (long long)blk * 1024;
    zp[i0] = (bf16)z0; zp[i1] = (bf16)z1; zp[i2] = (bf16)z2; zp[i3] = (bf16)z3;
    r1[tid] = z0 + z1 + z2 + z3;
    r2[tid] = z0 * z0 + z1 * z1 + z2 * z2 + z3 * z3;
    __syncthreads();
    for (int s = 128; s > 0; s >>= 1) {
        if (tid < s) { r1[tid] += r1[tid + s]; r2[tid] += r2[tid + s]; }
        __syncthreads();
    }
    if (tid == 0) {
        float mean = r1[0] * (1.0f / 1024.0f);
        float var = (r2[0] - 1024.0f * mean * mean) * (1.0f / 1023.0f);  // unbiased
        sfeat[b * 256 + d] = mean;
        sfeat[b * 256 + 128 + d] = sqrtf(fmaxf(var, 0.0f));
    }
}

// ---------------- f32 -> bf16 convert ----------------
__global__ void f32_to_bf16(const float* __restrict__ s, bf16* __restrict__ d, int n) {
    int i = blockIdx.x * 256 + threadIdx.x;
    if (i < n) d[i] = (bf16)s[i];
}

// ---------------- generic WMMA bf16 GEMM: C = A[M,K] * B[N,K]^T ----------------
// block tile 128x64, 8 waves of 32x32; async global->LDS DMA, double-buffered;
// optional bias / residual / relu; dual f32+bf16 out.
__global__ __launch_bounds__(256) void gemm_bf16_wmma(
    const bf16* __restrict__ A, const bf16* __restrict__ Bm,
    const float* __restrict__ bias, const float* __restrict__ residual,
    float* __restrict__ Cf, bf16* __restrict__ Cbf,
    int M, int N, int K, int relu,
    long long strA, long long strB, long long strC) {
    __shared__ __attribute__((aligned(64))) bf16 tA[2][128 * 32];
    __shared__ __attribute__((aligned(64))) bf16 tB[2][64 * 32];
    const int tid = threadIdx.x;
    const int lane = tid & 31;
    const int wave = tid >> 5;
    const int wm = (wave >> 1) * 32;   // 0..96
    const int wn = (wave & 1) * 32;    // 0,32
    const long long bz = blockIdx.z;
    const bf16* Ab = A + bz * strA;
    const bf16* Bb = Bm + bz * strB;
    const int bm = blockIdx.y * 128;
    const int bn = blockIdx.x * 64;

    v8f acc00 = {}, acc01 = {}, acc10 = {}, acc11 = {};

    const int mlo = lane & 15;
    const int kbA = (lane >> 4) * 16;  // A: lanes<16 -> K 0..15; else K 16..31
    const int kbB = (lane >> 4) * 16;  // B: lanes<16 -> K 0..15; else K 16..31

    // staging geometry: A tile 128x32 (2 chunks/thread), B tile 64x32 (1 chunk/thread)
    const int rowA = tid >> 2;               // 0..63
    const int seg  = (tid & 3) * 8;          // bf16 elements (16B chunks)

    // async DMA of one 32-wide K tile into buffer `buf`
    auto issue = [&](int kt, int buf) {
        {
            unsigned la = (unsigned)(size_t)(&tA[buf][rowA * 32 + seg]);
            const bf16* ga = Ab + (long long)(bm + rowA) * K + kt + seg;
            asm volatile("global_load_async_to_lds_b128 %0, %1, off"
                         :: "v"(la), "v"(ga) : "memory");
        }
        {
            unsigned la = (unsigned)(size_t)(&tA[buf][(rowA + 64) * 32 + seg]);
            const bf16* ga = Ab + (long long)(bm + rowA + 64) * K + kt + seg;
            asm volatile("global_load_async_to_lds_b128 %0, %1, off"
                         :: "v"(la), "v"(ga) : "memory");
        }
        {
            unsigned lb = (unsigned)(size_t)(&tB[buf][rowA * 32 + seg]);
            const bf16* gb = Bb + (long long)(bn + rowA) * K + kt + seg;
            asm volatile("global_load_async_to_lds_b128 %0, %1, off"
                         :: "v"(lb), "v"(gb) : "memory");
        }
    };

    const int nstep = K >> 5;
    issue(0, 0);
    for (int s = 0; s < nstep; ++s) {
        const int buf = s & 1;
        asm volatile("s_wait_asynccnt 0x0" ::: "memory");  // this step's DMA landed
        __syncthreads();                                   // visible to all waves
        if (s + 1 < nstep) issue((s + 1) << 5, buf ^ 1);   // DMA next tile under compute

        const bf16* sAb = tA[buf];
        const bf16* sBb = tB[buf];
        v8bf a0lo = *(const v8bf*)(&sAb[(wm + mlo) * 32 + kbA]);
        v8bf a0hi = *(const v8bf*)(&sAb[(wm + mlo) * 32 + kbA + 8]);
        v8bf a1lo = *(const v8bf*)(&sAb[(wm + 16 + mlo) * 32 + kbA]);
        v8bf a1hi = *(const v8bf*)(&sAb[(wm + 16 + mlo) * 32 + kbA + 8]);
        v16bf a0 = __builtin_shufflevector(a0lo, a0hi, 0,1,2,3,4,5,6,7,8,9,10,11,12,13,14,15);
        v16bf a1 = __builtin_shufflevector(a1lo, a1hi, 0,1,2,3,4,5,6,7,8,9,10,11,12,13,14,15);
        v16bf b0 = *(const v16bf*)(&sBb[(wn + mlo) * 32 + kbB]);
        v16bf b1 = *(const v16bf*)(&sBb[(wn + 16 + mlo) * 32 + kbB]);
        acc00 = __builtin_amdgcn_wmma_f32_16x16x32_bf16(false, a0, false, b0, (short)0, acc00, false, false);
        acc01 = __builtin_amdgcn_wmma_f32_16x16x32_bf16(false, a0, false, b1, (short)0, acc01, false, false);
        acc10 = __builtin_amdgcn_wmma_f32_16x16x32_bf16(false, a1, false, b0, (short)0, acc10, false, false);
        acc11 = __builtin_amdgcn_wmma_f32_16x16x32_bf16(false, a1, false, b1, (short)0, acc11, false, false);
    }

    const int cn = lane & 15;
    const int rbase = (lane >> 4) * 8;   // C: VGPR r -> row r (+8 for upper lanes)
    float* Cfb = Cf ? Cf + bz * strC : nullptr;
    bf16* Cbb = Cbf ? Cbf + bz * strC : nullptr;
    const float* Rb = residual ? residual + bz * strC : nullptr;

    auto epi = [&](v8f acc, int mi, int ni) {
        #pragma unroll
        for (int r = 0; r < 8; ++r) {
            int row = bm + wm + mi * 16 + rbase + r;
            int col = bn + wn + ni * 16 + cn;
            float v = acc[r];
            if (bias) v += bias[col];
            if (Rb)   v += Rb[(long long)row * N + col];
            if (relu) v = fmaxf(v, 0.0f);
            long long o = (long long)row * N + col;
            if (Cfb) Cfb[o] = v;
            if (Cbb) Cbb[o] = (bf16)v;
        }
    };
    epi(acc00, 0, 0); epi(acc01, 0, 1); epi(acc10, 1, 0); epi(acc11, 1, 1);
}

// ---------------- attention: scores + softmax ----------------
__global__ __launch_bounds__(128) void attn_scores(const float* __restrict__ qkv,
                                                   float* __restrict__ att) {
    __shared__ float sK[128 * 64];
    const int blk = blockIdx.x;   // b*8 + h
    const int b = blk >> 3, h = blk & 7;
    const int tid = threadIdx.x;  // query token
    for (int idx = tid; idx < 128 * 64; idx += 128) {
        int tok = idx >> 6, dd = idx & 63;
        sK[idx] = qkv[((long long)(b * 128 + tok)) * 1536 + 512 + h * 64 + dd];
    }
    __syncthreads();
    float q[64];
    const float* qp = qkv + ((long long)(b * 128 + tid)) * 1536 + h * 64;
    #pragma unroll
    for (int dd = 0; dd < 64; ++dd) q[dd] = qp[dd];
    float* arow = att + (((long long)blk) * 128 + tid) * 128;
    float mx = -1e30f;
    for (int k = 0; k < 128; ++k) {
        float s = 0.0f;
        #pragma unroll
        for (int dd = 0; dd < 64; ++dd) s += q[dd] * sK[k * 64 + dd];
        s *= 0.125f;                     // 1/sqrt(64)
        arow[k] = s;
        mx = fmaxf(mx, s);
    }
    float sum = 0.0f;
    for (int k = 0; k < 128; ++k) { float e = __expf(arow[k] - mx); arow[k] = e; sum += e; }
    float inv = 1.0f / sum;
    for (int k = 0; k < 128; ++k) arow[k] *= inv;
}

// ---------------- attention: ctx = att @ v  (writes bf16) ----------------
__global__ __launch_bounds__(256) void attn_ctx(const float* __restrict__ qkv,
                                                const float* __restrict__ att,
                                                bf16* __restrict__ ctxbf) {
    __shared__ float sV[128 * 64];
    const int blk = blockIdx.x;
    const int b = blk >> 3, h = blk & 7;
    const int tid = threadIdx.x;
    for (int idx = tid; idx < 128 * 64; idx += 256) {
        int tok = idx >> 6, dd = idx & 63;
        sV[idx] = qkv[((long long)(b * 128 + tok)) * 1536 + 1024 + h * 64 + dd];
    }
    __syncthreads();
    const int q = tid >> 1, db = (tid & 1) * 32;
    const float* arow = att + (((long long)blk) * 128 + q) * 128;
    float acc[32];
    #pragma unroll
    for (int dd = 0; dd < 32; ++dd) acc[dd] = 0.0f;
    for (int k = 0; k < 128; ++k) {
        float a = arow[k];
        #pragma unroll
        for (int dd = 0; dd < 32; ++dd) acc[dd] += a * sV[k * 64 + db + dd];
    }
    bf16* cp = ctxbf + ((long long)(b * 128 + q)) * 512 + h * 64 + db;
    #pragma unroll
    for (int dd = 0; dd < 32; ++dd) cp[dd] = (bf16)acc[dd];
}

// ---------------- LayerNorm over 512, writes f32 + bf16 ----------------
__global__ __launch_bounds__(256) void layernorm_rw(const float* __restrict__ in,
                                                    const float* __restrict__ g,
                                                    const float* __restrict__ bta,
                                                    float* __restrict__ outF,
                                                    bf16* __restrict__ outBf) {
    __shared__ float r1[256], r2[256];
    const long long row = blockIdx.x;
    const float* xr = in + row * 512;
    const int tid = threadIdx.x;
    float a = xr[tid], b = xr[tid + 256];
    r1[tid] = a + b; r2[tid] = a * a + b * b;
    __syncthreads();
    for (int s = 128; s > 0; s >>= 1) {
        if (tid < s) { r1[tid] += r1[tid + s]; r2[tid] += r2[tid + s]; }
        __syncthreads();
    }
    float mean = r1[0] * (1.0f / 512.0f);
    float var = r2[0] * (1.0f / 512.0f) - mean * mean;   // biased
    float inv = rsqrtf(var + 1e-5f);
    float v0 = (a - mean) * inv * g[tid] + bta[tid];
    float v1 = (b - mean) * inv * g[tid + 256] + bta[tid + 256];
    if (outF)  { outF[row * 512 + tid] = v0;       outF[row * 512 + tid + 256] = v1; }
    if (outBf) { outBf[row * 512 + tid] = (bf16)v0; outBf[row * 512 + tid + 256] = (bf16)v1; }
}

// ---------------- factor MLP ----------------
__global__ __launch_bounds__(64) void mlp1_kernel(const float* __restrict__ sfeat,
                                                  const float* __restrict__ w,
                                                  const float* __restrict__ bias,
                                                  float* __restrict__ h1) {
    __shared__ float srow[256];
    const int b = blockIdx.x, j = threadIdx.x;
    for (int k = j; k < 256; k += 64) srow[k] = sfeat[b * 256 + k];
    __syncthreads();
    float acc = bias[j];
    for (int k = 0; k < 256; ++k) acc += srow[k] * w[j * 256 + k];
    h1[b * 64 + j] = fmaxf(acc, 0.0f);
}

__global__ __launch_bounds__(32) void mlp2_kernel(const float* __restrict__ h1,
                                                  const float* __restrict__ w,
                                                  const float* __restrict__ bias,
                                                  float* __restrict__ h2) {
    const int b = blockIdx.x, j = threadIdx.x;
    float acc = bias[j];
    for (int k = 0; k < 64; ++k) acc += h1[b * 64 + k] * w[j * 64 + k];
    h2[b * 32 + j] = fmaxf(acc, 0.0f);
}

__global__ __launch_bounds__(256) void mlp3_kernel(const float* __restrict__ h2,
                                                   const float* __restrict__ w,
                                                   const float* __restrict__ bias,
                                                   float* __restrict__ lower) {
    __shared__ float sh[32];
    const int b = blockIdx.x;
    const int o = blockIdx.y * 256 + threadIdx.x;
    if (threadIdx.x < 32) sh[threadIdx.x] = h2[b * 32 + threadIdx.x];
    __syncthreads();
    float acc = bias[o];
    #pragma unroll
    for (int k = 0; k < 32; ++k) acc += sh[k] * w[o * 32 + k];
    lower[(long long)b * 16384 + o] = acc;
}

// lam[b] = lower[b] @ lower[b]^T  (fp32, written straight into d_out)
__global__ __launch_bounds__(128) void lam_kernel(const float* __restrict__ lower,
                                                  float* __restrict__ lam) {
    __shared__ float li[128];
    const int b = blockIdx.x, i = blockIdx.y, j = threadIdx.x;
    li[j] = lower[(long long)b * 16384 + i * 128 + j];
    __syncthreads();
    const float* lj = lower + (long long)b * 16384 + j * 128;
    float acc = 0.0f;
    #pragma unroll 8
    for (int m = 0; m < 128; ++m) acc += li[m] * lj[m];
    lam[(long long)b * 16384 + i * 128 + j] = acc;
}

// y[b,i] = sum_m g[b,i,m]*u[i,m] + head_bias[i]; broadcast over pred_len
__global__ __launch_bounds__(128) void head_kernel(const float* __restrict__ g,
                                                   const float* __restrict__ u,
                                                   const float* __restrict__ hb,
                                                   float* __restrict__ y, int pred) {
    const int b = blockIdx.x, i = threadIdx.x;
    const float* gr = g + (long long)b * 16384 + i * 128;
    const float* ur = u + i * 128;
    float acc = 0.0f;
    #pragma unroll 8
    for (int m = 0; m < 128; ++m) acc += gr[m] * ur[m];
    acc += hb[i];
    float* yp = y + ((long long)b * 128 + i) * (long long)pred;
    for (int p = 0; p < pred; ++p) yp[p] = acc;
}

__global__ void copy_f32(const float* __restrict__ s, float* __restrict__ d, int n) {
    int i = blockIdx.x * 256 + threadIdx.x;
    if (i < n) d[i] = s[i];
}

// ---------------- host orchestration ----------------
extern "C" void kernel_launch(void* const* d_in, const int* in_sizes, int n_in,
                              void* d_out, int out_size, void* d_ws, size_t ws_size,
                              hipStream_t stream) {
    (void)in_sizes; (void)n_in;
    if (ws_size < WS_NEED) return;

    const float* x      = (const float*)d_in[0];
    const float* emb_w  = (const float*)d_in[1];
    const float* emb_b  = (const float*)d_in[2];
    const float* qkv_w  = (const float*)d_in[3];
    const float* qkv_b  = (const float*)d_in[4];
    const float* out_w  = (const float*)d_in[5];
    const float* out_b  = (const float*)d_in[6];
    const float* ln1_g  = (const float*)d_in[7];
    const float* ln1_b  = (const float*)d_in[8];
    const float* ln2_g  = (const float*)d_in[9];
    const float* ln2_b  = (const float*)d_in[10];
    const float* ff1_w  = (const float*)d_in[11];
    const float* ff1_b  = (const float*)d_in[12];
    const float* ff2_w  = (const float*)d_in[13];
    const float* ff2_b  = (const float*)d_in[14];
    const float* fn_g   = (const float*)d_in[15];
    const float* fn_b   = (const float*)d_in[16];
    const float* mlp1_w = (const float*)d_in[17];
    const float* mlp1_b = (const float*)d_in[18];
    const float* mlp2_w = (const float*)d_in[19];
    const float* mlp2_b = (const float*)d_in[20];
    const float* mlp3_w = (const float*)d_in[21];
    const float* mlp3_b = (const float*)d_in[22];
    const float* proj_w = (const float*)d_in[23];
    const float* proj_b = (const float*)d_in[24];
    const float* u      = (const float*)d_in[25];
    const float* head_b = (const float*)d_in[26];

    char* ws = (char*)d_ws;
    float* qtab   = (float*)(ws + OFF_QT);
    float* sfeat  = (float*)(ws + OFF_S);
    float* h1     = (float*)(ws + OFF_H1);
    float* h2     = (float*)(ws + OFF_H2);
    float* lower  = (float*)(ws + OFF_LOWER);
    bf16*  ztbf   = (bf16*) (ws + OFF_ZTBF);
    bf16*  wemb   = (bf16*) (ws + OFF_WEMB);
    bf16*  wqkv   = (bf16*) (ws + OFF_WQKV);
    bf16*  wout   = (bf16*) (ws + OFF_WOUT);
    bf16*  wff1   = (bf16*) (ws + OFF_WFF1);
    bf16*  wff2   = (bf16*) (ws + OFF_WFF2);
    bf16*  wprj   = (bf16*) (ws + OFF_WPRJ);
    float* tbuf   = (float*)(ws + OFF_T);
    bf16*  tbf    = (bf16*) (ws + OFF_TBF);
    float* qkvbuf = (float*)(ws + OFF_QKV);
    float* prebuf = (float*)(ws + OFF_PRE);
    float* att    = (float*)(ws + OFF_ATT);
    bf16*  ffbf   = (bf16*) (ws + OFF_FFBF);
    bf16*  ctxbf  = (bf16*) (ws + OFF_CTXBF);
    float* hbuf   = (float*)(ws + OFF_PRE);     // reuse (qkv/pre region free post-loop)
    bf16*  hbf    = (bf16*) (ws + OFF_CTXBF);   // reuse ctx region
    float* hp     = (float*)(ws + OFF_HP);
    bf16*  hpbf   = (bf16*) (ws + OFF_HPBF);
    bf16*  lambf  = (bf16*) (ws + OFF_LAMBF);
    float* gbuf   = (float*)(ws + OFF_G);

    float* y_out   = (float*)d_out;
    int pred = (out_size - (64 * 128 * 128 + 128 * 128)) / (64 * 128);
    if (pred <= 0) pred = 96;
    float* lam_out = y_out + (long long)64 * 128 * pred;
    float* u_out   = lam_out + 64 * 128 * 128;

    auto gemm = [&](const bf16* A, const bf16* Bm, const float* bias, const float* res,
                    float* Cf, bf16* Cbf, int M, int N, int K, int relu, int batch,
                    long long sa, long long sb, long long sc) {
        dim3 g(N / 64, M / 128, batch), blk(256, 1, 1);
        gemm_bf16_wmma<<<g, blk, 0, stream>>>(A, Bm, bias, res, Cf, Cbf, M, N, K, relu, sa, sb, sc);
    };
    auto cvt = [&](const float* s, bf16* d, int n) {
        f32_to_bf16<<<(n + 255) / 256, 256, 0, stream>>>(s, d, n);
    };

    // 1) copula quantile table + rank/z + mean/std features
    build_qtab<<<4, 256, 0, stream>>>(qtab);
    copula_rank<<<8192, 256, 0, stream>>>(x, qtab, ztbf, sfeat);

    // 2) weight conversions to bf16
    cvt(emb_w, wemb, 512 * 1024);
    cvt(qkv_w, wqkv, 4 * 1536 * 512);
    cvt(out_w, wout, 4 * 512 * 512);
    cvt(ff1_w, wff1, 4 * 1024 * 512);
    cvt(ff2_w, wff2, 4 * 512 * 1024);
    cvt(proj_w, wprj, 128 * 512);

    // 3) factor MLP + lam (fp32, straight to output) + bf16 copy of lam
    mlp1_kernel<<<64, 64, 0, stream>>>(sfeat, mlp1_w, mlp1_b, h1);
    mlp2_kernel<<<64, 32, 0, stream>>>(h1, mlp2_w, mlp2_b, h2);
    mlp3_kernel<<<dim3(64, 64), 256, 0, stream>>>(h2, mlp3_w, mlp3_b, lower);
    lam_kernel<<<dim3(64, 128), 128, 0, stream>>>(lower, lam_out);
    cvt(lam_out, lambf, 64 * 128 * 128);

    // 4) embedding: t = z^T @ emb_w^T + emb_b   [8192,512]
    gemm(ztbf, wemb, emb_b, nullptr, tbuf, tbf, 8192, 512, 1024, 0, 1, 0, 0, 0);

    // 5) transformer layers
    for (int l = 0; l < 4; ++l) {
        gemm(tbf, wqkv + (long long)l * 1536 * 512, qkv_b + l * 1536, nullptr,
             qkvbuf, nullptr, 8192, 1536, 512, 0, 1, 0, 0, 0);
        attn_scores<<<512, 128, 0, stream>>>(qkvbuf, att);
        attn_ctx<<<512, 256, 0, stream>>>(qkvbuf, att, ctxbf);
        gemm(ctxbf, wout + (long long)l * 512 * 512, out_b + l * 512, tbuf,
             prebuf, nullptr, 8192, 512, 512, 0, 1, 0, 0, 0);
        layernorm_rw<<<8192, 256, 0, stream>>>(prebuf, ln1_g + l * 512, ln1_b + l * 512, tbuf, tbf);
        gemm(tbf, wff1 + (long long)l * 1024 * 512, ff1_b + l * 1024, nullptr,
             nullptr, ffbf, 8192, 1024, 512, 1, 1, 0, 0, 0);
        gemm(ffbf, wff2 + (long long)l * 512 * 1024, ff2_b + l * 512, tbuf,
             prebuf, nullptr, 8192, 512, 1024, 0, 1, 0, 0, 0);
        layernorm_rw<<<8192, 256, 0, stream>>>(prebuf, ln2_g + l * 512, ln2_b + l * 512, tbuf, tbf);
    }

    // 6) final norm, projection head
    layernorm_rw<<<8192, 256, 0, stream>>>(tbuf, fn_g, fn_b, hbuf, hbf);
    gemm(hbf, wprj, proj_b, nullptr, hp, hpbf, 8192, 128, 512, 0, 1, 0, 0, 0);

    // 7) g = hp @ lam (per batch; lam symmetric so B operand is K-contiguous)
    gemm(hpbf, lambf, nullptr, nullptr, gbuf, nullptr, 128, 128, 128, 0, 64,
         16384, 16384, 16384);

    // 8) outputs: y (broadcast pred_len), u copy
    head_kernel<<<64, 128, 0, stream>>>(gbuf, u, head_b, y_out, pred);
    copy_f32<<<64, 256, 0, stream>>>(u, u_out, 128 * 128);
}
